// Attention_21354577395789
// MI455X (gfx1250) — matrix operand order repaired and checked by reading, not compiled, not run on hardware
//
#include <hip/hip_runtime.h>
#include <hip/hip_bf16.h>
#include <math.h>

typedef __attribute__((ext_vector_type(16))) _Float16 v16h;
typedef __attribute__((ext_vector_type(8)))  _Float16 v8h;
typedef __attribute__((ext_vector_type(4)))  _Float16 v4h;
typedef __attribute__((ext_vector_type(8)))  float    v8f;
typedef __attribute__((ext_vector_type(4)))  float    v4f;

// Problem sizes (fixed by the reference)
constexpr int Hn   = 1024;          // hidden size H (= N of big GEMM)
constexpr int Bn   = 16;
constexpr int Sn   = 2048;
constexpr int Kn   = 2048;          // 2H (= K of big GEMM)
constexpr int Mn   = Bn * Sn;       // 32768 rows
constexpr int KC   = Kn / 32;       // 64 k-chunks of 32
constexpr int NT   = Hn / 16;       // 64 n-tiles of 16

// LDS layout for energy_kernel (dynamic shared)
constexpr int A_STRIDE   = 2048 + 8;                 // halves, +16B pad to avoid bank conflicts
constexpr int SA_BYTES   = 32 * A_STRIDE * 2;        // 131584
constexpr int SRED_BYTES = 256 * 8 * 4;              // 8192
constexpr int LDS_BYTES  = SA_BYTES + SRED_BYTES;    // 139776

// Workspace layout (bytes)
constexpr size_t WS_PACKB  = 0;                        // 64*64*32*16 halves = 4 MB
constexpr size_t WS_BIAS   = 4194304;                  // 1024 f32
constexpr size_t WS_SCORES = WS_BIAS + 4096;           // 32768 f32
constexpr size_t WS_ATTN   = WS_SCORES + Mn * 4;       // 32768 f32

// ---------------------------------------------------------------------------
// Kernel 0a: bias'[n] = attn_b[n] + sum_j hidden[j] * attn_w[n, j]  (j < 2H)
// ---------------------------------------------------------------------------
__global__ __launch_bounds__(256) void bias_kernel(
    const float* __restrict__ hidden, const float* __restrict__ attn_w,
    const float* __restrict__ attn_b, float* __restrict__ biasP) {
  int n = blockIdx.x * 256 + threadIdx.x;            // 4 blocks -> 1024
  const v4f* w4 = (const v4f*)(attn_w + (size_t)n * 4096);
  const v4f* h4 = (const v4f*)hidden;
  float acc = 0.f;
#pragma unroll 8
  for (int j = 0; j < 512; ++j) {
    v4f a = h4[j], b = w4[j];
    acc += a.x * b.x + a.y * b.y + a.z * b.z + a.w * b.w;
  }
  biasP[n] = attn_b[n] + acc;
}

// ---------------------------------------------------------------------------
// Kernel 0b: pack W_e = attn_w[:, 2H:4H] into f16 WMMA B-fragment layout.
// B element (k,n) = attn_w[n*4H + 2H + k].
// Fragment (nt,kc): lane L holds halves h=0..15 with
//   N = nt*16 + (L&15),  K = kc*32 + ((L>>4)<<4) + h      (contiguous k order, same as A)
// Stored at halves[((nt*KC+kc)*32 + L)*16 + h]  -> each lane owns 32 aligned bytes.
// ---------------------------------------------------------------------------
__global__ __launch_bounds__(32) void pack_kernel(
    const float* __restrict__ attn_w, _Float16* __restrict__ packedB) {
  int blk = blockIdx.x;                               // 4096 = nt*KC + kc
  int L   = threadIdx.x;                              // 0..31
  int nt  = blk >> 6, kc = blk & 63;
  int n     = nt * 16 + (L & 15);
  int kbase = kc * 32 + ((L >> 4) << 4);
  const v4f* src = (const v4f*)(attn_w + (size_t)n * 4096 + 2048 + kbase);
  v16h out;
#pragma unroll
  for (int q = 0; q < 4; ++q) {
    v4f w = src[q];
    out[q * 4 + 0] = (_Float16)w.x;
    out[q * 4 + 1] = (_Float16)w.y;
    out[q * 4 + 2] = (_Float16)w.z;
    out[q * 4 + 3] = (_Float16)w.w;
  }
  *(v16h*)(packedB + ((size_t)blk * 32 + L) * 16) = out;
}

// ---------------------------------------------------------------------------
// Kernel 1: fused  scores[m] = sum_n tanh( (enc @ W_e^T)[m,n] + bias'[n] ) * v[n]
// Block = 256 threads (8 waves), 32 rows per block, full N=1024, K=2048.
// A (enc rows) converted fp32->f16 once into LDS; B fragments stream from L2.
// ---------------------------------------------------------------------------
__global__ __launch_bounds__(256) void energy_kernel(
    const float* __restrict__ enc, const _Float16* __restrict__ packedB,
    const float* __restrict__ biasP, const float* __restrict__ vP,
    float* __restrict__ scores) {
  extern __shared__ char smem[];
  _Float16* sA  = (_Float16*)smem;                    // [32][A_STRIDE] halves
  float*    sRed = (float*)(smem + SA_BYTES);         // [256][8]

  const int tid = threadIdx.x;
  const int m0  = blockIdx.x * 32;

  // ---- Stage A: 32 rows x 2048 fp32 -> f16 LDS (coalesced b128 loads) ----
#pragma unroll 4
  for (int it = 0; it < 64; ++it) {
    int flat = it * 256 + tid;                        // float4 index, 0..16383
    int row  = flat >> 9;                             // /512
    int c4   = flat & 511;
    v4f d = *(const v4f*)(enc + ((size_t)(m0 + row)) * 2048 + (size_t)c4 * 4);
    v4h hd = {(_Float16)d.x, (_Float16)d.y, (_Float16)d.z, (_Float16)d.w};
    *(v4h*)(sA + row * A_STRIDE + c4 * 4) = hd;
  }
  __syncthreads();

  // ---- WMMA main loop ----
  const int w      = tid >> 5;                        // wave 0..7
  const int L      = tid & 31;
  const int rowSel = w & 1;                           // which 16-row fragment
  const int g      = w >> 1;                          // n-tile group 0..3

  // A fragment (16-bit A 16x32, contiguous k order): lane L -> row (L&15);
// halves 0..15 = K kbase..kbase+15,
// kbase = kc*32 + 16*(L>>4).
  const _Float16* aRow = sA + (size_t)(16 * rowSel + (L & 15)) * A_STRIDE + 16 * (L >> 4);

  float acc[8];
#pragma unroll
  for (int r = 0; r < 8; ++r) acc[r] = 0.f;

  for (int j = 0; j < 8; ++j) {                       // 16 n-tiles per wave, 2 at a time
    int nt0 = j * 8 + g * 2;
    int nt1 = nt0 + 1;
    v8f c0 = {}; v8f c1 = {};
    const v16h* b0p = (const v16h*)packedB + (size_t)nt0 * KC * 32 + L;
    const v16h* b1p = (const v16h*)packedB + (size_t)nt1 * KC * 32 + L;
#pragma unroll 4
    for (int kc = 0; kc < KC; ++kc) {
      const _Float16* ap = aRow + kc * 32;
      v8h lo = *(const v8h*)ap;                       // ds_load_b128
      v8h hi = *(const v8h*)(ap + 8);                 // ds_load_b128
      v16h a = __builtin_shufflevector(lo, hi, 0,1,2,3,4,5,6,7,8,9,10,11,12,13,14,15);
      v16h b0 = b0p[(size_t)kc * 32];
      v16h b1 = b1p[(size_t)kc * 32];
      if (kc + 1 < KC) __builtin_prefetch(b0p + (size_t)(kc + 1) * 32, 0, 1);
      c0 = __builtin_amdgcn_wmma_f32_16x16x32_f16(false, a, false, b0, (short)0, c0, false, false);
      c1 = __builtin_amdgcn_wmma_f32_16x16x32_f16(false, a, false, b1, (short)0, c1, false, false);
    }
    // Fused epilogue: energy = tanh(C + bias'), partial score += energy * v
    // C/D layout: lane L, vgpr r -> row r + 8*(L>>4), col (L&15) within tile.
    int n0 = nt0 * 16 + (L & 15);
    int n1 = nt1 * 16 + (L & 15);
    float bn0 = biasP[n0], vn0 = vP[n0];
    float bn1 = biasP[n1], vn1 = vP[n1];
#pragma unroll
    for (int r = 0; r < 8; ++r)
      acc[r] += tanhf(c0[r] + bn0) * vn0 + tanhf(c1[r] + bn1) * vn1;
  }

  // ---- Deterministic fixed-order reduction to per-row scores ----
#pragma unroll
  for (int r = 0; r < 8; ++r) sRed[tid * 8 + r] = acc[r];
  __syncthreads();

  if (tid < 32) {
    int t       = tid;                                // local row 0..31
    int wlow    = t >> 4;                             // waves with rowSel == wlow
    int lr      = t & 15;
    int halfSel = (lr >= 8) ? 1 : 0;                  // lanes 16..31 hold rows +8
    int r       = lr - 8 * halfSel;
    float s = 0.f;
    for (int wg = 0; wg < 4; ++wg) {
      int wv = wg * 2 + wlow;
      for (int lc = 0; lc < 16; ++lc) {
        int lane = halfSel * 16 + lc;
        s += sRed[(wv * 32 + lane) * 8 + r];
      }
    }
    scores[m0 + t] = s;
  }
}

// ---------------------------------------------------------------------------
// Kernel 2: softmax over S per batch (deterministic LDS tree reductions)
// ---------------------------------------------------------------------------
__global__ __launch_bounds__(256) void softmax_kernel(
    const float* __restrict__ scores, float* __restrict__ attn) {
  __shared__ float red[256];
  int b = blockIdx.x, tid = threadIdx.x;
  const float* sc = scores + (size_t)b * Sn;
  float vloc[8];
  float mx = -INFINITY;
#pragma unroll
  for (int i = 0; i < 8; ++i) { vloc[i] = sc[i * 256 + tid]; mx = fmaxf(mx, vloc[i]); }
  red[tid] = mx; __syncthreads();
  for (int st = 128; st > 0; st >>= 1) {
    if (tid < st) red[tid] = fmaxf(red[tid], red[tid + st]);
    __syncthreads();
  }
  mx = red[0]; __syncthreads();
  float sum = 0.f;
#pragma unroll
  for (int i = 0; i < 8; ++i) { vloc[i] = __expf(vloc[i] - mx); sum += vloc[i]; }
  red[tid] = sum; __syncthreads();
  for (int st = 128; st > 0; st >>= 1) {
    if (tid < st) red[tid] += red[tid + st];
    __syncthreads();
  }
  float inv = 1.0f / red[0];
#pragma unroll
  for (int i = 0; i < 8; ++i) attn[(size_t)b * Sn + i * 256 + tid] = vloc[i] * inv;
}

// ---------------------------------------------------------------------------
// Kernel 3: context[b,d] = sum_s attn[b,s] * enc[b,s,d]   (memory-bound pass)
// ---------------------------------------------------------------------------
__global__ __launch_bounds__(256) void context_kernel(
    const float* __restrict__ enc, const float* __restrict__ attn,
    float* __restrict__ out) {
  int b = blockIdx.x >> 3;                            // 128 blocks
  int d = (blockIdx.x & 7) * 256 + threadIdx.x;       // 0..2047
  const float* ep = enc + (size_t)b * Sn * Kn + d;
  const float* ap = attn + (size_t)b * Sn;
  float acc = 0.f;
#pragma unroll 8
  for (int s = 0; s < Sn; ++s) acc += ap[s] * ep[(size_t)s * Kn];
  out[(size_t)b * Kn + d] = acc;
}

// ---------------------------------------------------------------------------
extern "C" void kernel_launch(void* const* d_in, const int* in_sizes, int n_in,
                              void* d_out, int out_size, void* d_ws, size_t ws_size,
                              hipStream_t stream) {
  (void)in_sizes; (void)n_in; (void)out_size; (void)ws_size;
  const float* hidden = (const float*)d_in[0];
  const float* enc    = (const float*)d_in[1];
  const float* attn_w = (const float*)d_in[2];
  const float* attn_b = (const float*)d_in[3];
  const float* v_w    = (const float*)d_in[4];
  float* out = (float*)d_out;

  char* ws = (char*)d_ws;
  _Float16* packedB = (_Float16*)(ws + WS_PACKB);
  float*    biasP   = (float*)(ws + WS_BIAS);
  float*    scores  = (float*)(ws + WS_SCORES);
  float*    attn    = (float*)(ws + WS_ATTN);

  static int lds_attr_set = 0;  // idempotent, does not affect determinism of work
  if (!lds_attr_set) {
    hipFuncSetAttribute((const void*)energy_kernel,
                        hipFuncAttributeMaxDynamicSharedMemorySize, LDS_BYTES);
    lds_attr_set = 1;
  }

  bias_kernel<<<dim3(4), dim3(256), 0, stream>>>(hidden, attn_w, attn_b, biasP);
  pack_kernel<<<dim3(NT * KC), dim3(32), 0, stream>>>(attn_w, packedB);
  energy_kernel<<<dim3(Mn / 32), dim3(256), LDS_BYTES, stream>>>(enc, packedB, biasP, v_w, scores);
  softmax_kernel<<<dim3(Bn), dim3(256), 0, stream>>>(scores, attn);
  context_kernel<<<dim3(Bn * 8), dim3(256), 0, stream>>>(enc, attn, out);
}
